// MultiHeadAttentionBase_60060822667814
// MI455X (gfx1250) — compile-verified, matrix operand order checked by reading, not run on hardware
//
#include <hip/hip_runtime.h>

#define B_ 4
#define S_ 2048
#define E_ 1024
#define H_ 16
#define D_ 64
#define KT 64
#define NT (S_ / KT)

static constexpr float NEG2   = -1.4426950408889634e9f;   // -1e9 * log2(e)
static constexpr float QSCALE = 0.18033688011112042f;     // (1/sqrt(64)) * log2(e)

typedef __attribute__((ext_vector_type(16))) __bf16 v16bf;
typedef __attribute__((ext_vector_type(8)))  __bf16 v8bf;
typedef __attribute__((ext_vector_type(8)))  float  v8f;

// Flash-attention: one block = one (b, h, 128-query tile); 8 waves x 16 rows.
// 64-key tiles, bf16 WMMA (f32 accum), online softmax in exp2 domain.
__global__ __launch_bounds__(256) void fa_wmma_kernel(
    const float* __restrict__ Q, const float* __restrict__ K,
    const float* __restrict__ V, const int* __restrict__ M,
    float* __restrict__ O)
{
    __shared__ __bf16 k_lds[KT][72];      // [key][feat]
    __shared__ __bf16 vt_lds[64][72];     // [feat][key] (transposed V)
    __shared__ __bf16 p_lds[8][16][72];   // per-wave P scratch [row][key 0..63]

    const int tid  = threadIdx.x;
    const int wave = tid >> 5;
    const int lane = tid & 31;
    const int lm   = lane & 15;
    const int half = lane >> 4;

    const int blk   = blockIdx.x;
    const int qtile = blk % (S_ / 128);
    const int h     = (blk / (S_ / 128)) % H_;
    const int b     = blk / ((S_ / 128) * H_);

    const int q0 = qtile * 128 + wave * 16;

    const float* Qb = Q + ((size_t)b * S_) * E_ + (size_t)h * D_;
    const float* Kb = K + ((size_t)b * S_) * E_ + (size_t)h * D_;
    const float* Vb = V + ((size_t)b * S_) * E_ + (size_t)h * D_;
    const int*   Mb = M + ((size_t)b * S_) * S_;

    // ---- resident Q A-fragments (2 chunks of 16x32), scale*log2e folded ----
    v16bf qa[2];
    {
        const float* qrow = Qb + (size_t)(q0 + lm) * E_;
        #pragma unroll
        for (int c = 0; c < 2; ++c) {
            #pragma unroll
            for (int e = 0; e < 8; ++e) {
                qa[c][e]     = (__bf16)(qrow[c * 32 + half * 8 + e] * QSCALE);
                qa[c][8 + e] = (__bf16)(qrow[c * 32 + 16 + half * 8 + e] * QSCALE);
            }
        }
    }

    v8f o[4];
    #pragma unroll
    for (int dc = 0; dc < 4; ++dc) o[dc] = (v8f){0.f,0.f,0.f,0.f,0.f,0.f,0.f,0.f};
    float rmax[8], rsum[8];
    #pragma unroll
    for (int v = 0; v < 8; ++v) { rmax[v] = -3.0e38f; rsum[v] = 0.0f; }

    // ---- staging / prefetch (per thread: 16 K elems + 16 V elems) ----
    const int sr  = tid >> 2;          // key row 0..63
    const int scg = (tid & 3) * 16;    // feature base
    float4 kf[4], vf[4];

    {   // prefetch tile 0
        const float* krow = Kb + (size_t)sr * E_ + scg;
        const float* vrow = Vb + (size_t)sr * E_ + scg;
        #pragma unroll
        for (int i = 0; i < 4; ++i) {
            kf[i] = *(const float4*)(krow + 4 * i);
            vf[i] = *(const float4*)(vrow + 4 * i);
        }
    }

    for (int kt = 0; kt < NT; ++kt) {
        __syncthreads();
        // ---- commit staged tile to LDS (bf16) ----
        {
            float ka[16], va[16];
            #pragma unroll
            for (int i = 0; i < 4; ++i)
                #pragma unroll
                for (int j = 0; j < 4; ++j) {
                    ka[4 * i + j] = ((const float*)&kf[i])[j];
                    va[4 * i + j] = ((const float*)&vf[i])[j];
                }
            v8bf p0, p1;
            #pragma unroll
            for (int j = 0; j < 8; ++j) { p0[j] = (__bf16)ka[j]; p1[j] = (__bf16)ka[8 + j]; }
            *(v8bf*)&k_lds[sr][scg]     = p0;
            *(v8bf*)&k_lds[sr][scg + 8] = p1;
            #pragma unroll
            for (int e = 0; e < 16; ++e) vt_lds[scg + e][sr] = (__bf16)va[e];
        }
        __syncthreads();

        // ---- prefetch next tile while computing this one ----
        if (kt + 1 < NT) {
            const float* krow = Kb + (size_t)((kt + 1) * KT + sr) * E_ + scg;
            const float* vrow = Vb + (size_t)((kt + 1) * KT + sr) * E_ + scg;
            #pragma unroll
            for (int i = 0; i < 4; ++i) {
                kf[i] = *(const float4*)(krow + 4 * i);
                vf[i] = *(const float4*)(vrow + 4 * i);
            }
        }

        // ---- GEMM1: scores for four 16-key subtiles ----
        v8f p[4];
        #pragma unroll
        for (int t = 0; t < 4; ++t) {
            v8f s = (v8f){0.f,0.f,0.f,0.f,0.f,0.f,0.f,0.f};
            const __bf16* kr = &k_lds[t * 16 + lm][0];
            #pragma unroll
            for (int c = 0; c < 2; ++c) {
                v16bf kb;
                #pragma unroll
                for (int e = 0; e < 8; ++e) {
                    kb[e]     = kr[c * 32 + half * 8 + e];
                    kb[8 + e] = kr[c * 32 + 16 + half * 8 + e];
                }
                s = __builtin_amdgcn_wmma_f32_16x16x32_bf16(
                        false, qa[c], false, kb, (short)0, s, false, false);
            }
            // mask on D fragment: row = v + 8*half, col = lane&15
            const int keyg = kt * KT + t * 16 + lm;
            #pragma unroll
            for (int v = 0; v < 8; ++v) {
                const int qr = q0 + v + half * 8;
                const int mv = Mb[(size_t)qr * S_ + keyg];
                s[v] = (mv == 0) ? NEG2 : s[v];
            }
            p[t] = s;
        }

        // ---- online softmax (exp2 domain); rsum kept as per-lane partial ----
        float sc_[8];
        #pragma unroll
        for (int v = 0; v < 8; ++v) {
            float tmax = fmaxf(fmaxf(p[0][v], p[1][v]), fmaxf(p[2][v], p[3][v]));
            #pragma unroll
            for (int off = 1; off < 16; off <<= 1)
                tmax = fmaxf(tmax, __shfl_xor(tmax, off, 32));
            const float nmax = fmaxf(rmax[v], tmax);
            const float corr = __builtin_amdgcn_exp2f(rmax[v] - nmax);
            rmax[v] = nmax;
            float ps = 0.0f;
            #pragma unroll
            for (int t = 0; t < 4; ++t) {
                const float e = __builtin_amdgcn_exp2f(p[t][v] - nmax);
                p[t][v] = e;
                ps += e;
            }
            rsum[v] = rsum[v] * corr + ps;   // per-lane partial (reduced at end)
            sc_[v] = corr;
        }

        // ---- P -> per-wave LDS scratch (D-layout -> row-major) ----
        #pragma unroll
        for (int v = 0; v < 8; ++v)
            #pragma unroll
            for (int t = 0; t < 4; ++t)
                p_lds[wave][v + half * 8][t * 16 + lm] = (__bf16)p[t][v];

        // ---- rescale accumulators ----
        #pragma unroll
        for (int dc = 0; dc < 4; ++dc)
            #pragma unroll
            for (int v = 0; v < 8; ++v) o[dc][v] *= sc_[v];

        // ---- GEMM2: O += P(16x64) * V(64x64) ----
        v16bf pa[2];
        {
            const __bf16* pr = &p_lds[wave][lm][0];
            #pragma unroll
            for (int c = 0; c < 2; ++c)
                #pragma unroll
                for (int e = 0; e < 8; ++e) {
                    pa[c][e]     = pr[c * 32 + half * 8 + e];
                    pa[c][8 + e] = pr[c * 32 + 16 + half * 8 + e];
                }
        }
        #pragma unroll
        for (int dc = 0; dc < 4; ++dc) {
            v8f acc = o[dc];
            const __bf16* vr = &vt_lds[dc * 16 + lm][0];
            #pragma unroll
            for (int c = 0; c < 2; ++c) {
                v16bf vb;
                #pragma unroll
                for (int e = 0; e < 8; ++e) {
                    vb[e]     = vr[c * 32 + half * 8 + e];
                    vb[8 + e] = vr[c * 32 + 16 + half * 8 + e];
                }
                acc = __builtin_amdgcn_wmma_f32_16x16x32_bf16(
                          false, pa[c], false, vb, (short)0, acc, false, false);
            }
            o[dc] = acc;
        }
    }

    // ---- final cross-lane row-sum reduction, normalize, store ----
    #pragma unroll
    for (int v = 0; v < 8; ++v) {
        float rs = rsum[v];
        #pragma unroll
        for (int off = 1; off < 16; off <<= 1)
            rs += __shfl_xor(rs, off, 32);
        const float inv = 1.0f / rs;
        const int qr = q0 + v + half * 8;
        float* orow = O + ((size_t)b * S_ + qr) * E_ + (size_t)h * D_;
        #pragma unroll
        for (int dc = 0; dc < 4; ++dc)
            orow[dc * 16 + lm] = o[dc][v] * inv;
    }
}

extern "C" void kernel_launch(void* const* d_in, const int* in_sizes, int n_in,
                              void* d_out, int out_size, void* d_ws, size_t ws_size,
                              hipStream_t stream) {
    const float* q = (const float*)d_in[0];
    const float* k = (const float*)d_in[1];
    const float* v = (const float*)d_in[2];
    const int*   m = (const int*)d_in[3];
    float* out = (float*)d_out;

    dim3 grid(B_ * H_ * (S_ / 128));
    dim3 block(256);
    fa_wmma_kernel<<<grid, block, 0, stream>>>(q, k, v, m, out);
}
